// Attention_11321533792474
// MI455X (gfx1250) — compile-verified, matrix operand order checked by reading, not run on hardware
//
#include <hip/hip_runtime.h>
#include <hip/hip_bf16.h>

// Problem sizes (fixed by the reference).
#define B_ 64
#define T_ 512
#define D_ 1024
#define U_ 1024

// Tiling for the fused scores GEMM.
#define TM 32          // T rows per workgroup (2 wave-rows of 16)
#define TN 128         // U columns per N-strip (4 wave-cols x 2 x 16)
#define TK 32          // K per WMMA step
#define A_STRIDE 1048  // halves; 2096 B row stride: 8/16B aligned, bank-staggered
#define B_STRIDE 40    // halves; 80 B row stride: 16B aligned, bank-staggered
#define NTHREADS 256

typedef __attribute__((ext_vector_type(16))) __bf16          v16bf;
typedef __attribute__((ext_vector_type(8)))  float           v8f;
typedef __attribute__((ext_vector_type(8)))  unsigned short  v8us;
typedef __attribute__((ext_vector_type(2)))  float           v2f;
typedef __attribute__((ext_vector_type(2)))  __bf16          v2bf;

union Frag {
    struct { v8us lo, hi; } p;
    v16bf v;
};

// Two fp32 -> packed bf16x2 in one v_cvt_pk_bf16_f32.
__device__ __forceinline__ unsigned pack2bf(float a, float b) {
    v2f f = {a, b};
    union { v2bf h; unsigned u; } r;
    r.h = __builtin_convertvector(f, v2bf);
    return r.u;
}

// ---------------------------------------------------------------------------
// Kernel 1: whbm[b,u] = sum_d ht[b,d] * Wh[d,u] + bh[u] + bm[u]
// (bm folded in; bv dropped entirely: softmax is shift-invariant.)
// ---------------------------------------------------------------------------
__global__ __launch_bounds__(NTHREADS) void whbm_kernel(
    const float* __restrict__ ht, const float* __restrict__ Wh,
    const float* __restrict__ bh, const float* __restrict__ bm,
    float* __restrict__ whbm)
{
    const int b = blockIdx.y;
    const int u = blockIdx.x * NTHREADS + threadIdx.x;
    float acc = bh[u] + bm[u];
    const float* h = ht + (size_t)b * D_;
    for (int d = 0; d < D_; ++d)
        acc += h[d] * Wh[(size_t)d * U_ + u];
    whbm[b * U_ + u] = acc;
}

// ---------------------------------------------------------------------------
// Kernel 2 (main): fused  scores[b,t] = sum_u tanh(mt@Wm + whbm) * Wv
// One workgroup = (batch b, 32 T-rows). 8 waves = 2(M) x 4(N); each wave owns
// TWO 16x16 WMMA tiles in a 128-wide U strip (A fragment reused, 2 WMMAs per
// barrier). mt tile resident in LDS at full K; Wm double-buffered per 32-K
// step so staging overlaps WMMA and only ONE barrier per k-step is needed.
// The [B,T,U] intermediate never leaves registers.
// ---------------------------------------------------------------------------
__global__ __launch_bounds__(NTHREADS) void bahdanau_scores_kernel(
    const float* __restrict__ mt, const float* __restrict__ Wm,
    const float* __restrict__ whbm, const float* __restrict__ Wv,
    float* __restrict__ scores)
{
    __shared__ unsigned short lA[TM * A_STRIDE];       // mt tile [t][k] bf16 (~67 KB)
    __shared__ unsigned short lB[2][TN * B_STRIDE];    // Wm tile [u][k] bf16, dbl-buffered
    __shared__ float lds_score[TM];

    const int tid   = threadIdx.x;
    const int b     = blockIdx.y;
    const int t0    = blockIdx.x * TM;
    const int lane  = tid & 31;
    const int wave  = tid >> 5;
    const int waveM = wave >> 2;      // 0..1 -> which 16-row M tile
    const int waveN = wave & 3;       // 0..3 -> which 32-wide N sub-strip

    // --- Stage full-K A tile once: mt[b, t0..t0+31, 0..1023] -> bf16 LDS ---
    // float4 global loads, two packed cvts, one ds_store_b64 per 4 elements.
    const float* mtb = mt + ((size_t)(b * T_ + t0)) * D_;
    #pragma unroll 4
    for (int j = 0; j < (TM * D_ / 4) / NTHREADS; ++j) {   // 32 fixed iters
        int i  = j * NTHREADS + tid;
        int t  = i >> 8;            // D_/4 = 256 float4 per row
        int k4 = i & 255;
        float4 f = *(const float4*)&mtb[(size_t)t * D_ + 4 * k4];
        uint2 p; p.x = pack2bf(f.x, f.y); p.y = pack2bf(f.z, f.w);
        *(uint2*)&lA[t * A_STRIDE + 4 * k4] = p;
    }

    // Stage Wm[k0..k0+31][n0..n0+127] transposed into lB[buf][u][k]:
    // float4 loads along contiguous u, 4 packed ds_store_b32 scatters.
    auto stageB = [&](int buf, int k0, int n0) {
        #pragma unroll
        for (int j = 0; j < ((TK / 2) * (TN / 4)) / NTHREADS; ++j) {  // 2 fixed iters
            int i  = j * NTHREADS + tid;
            int u4 = i & 31;            // TN/4 = 32 float4 columns
            int k2 = i >> 5;            // 16 row-pairs
            int u  = u4 * 4;
            size_t g = (size_t)(k0 + 2 * k2) * U_ + (n0 + u);
            float4 f0 = *(const float4*)&Wm[g];        // row k0+2k2
            float4 f1 = *(const float4*)&Wm[g + U_];   // row k0+2k2+1
            unsigned short* dst = &lB[buf][u * B_STRIDE + 2 * k2];
            *(unsigned*)(dst)                = pack2bf(f0.x, f1.x);
            *(unsigned*)(dst + B_STRIDE)     = pack2bf(f0.y, f1.y);
            *(unsigned*)(dst + 2 * B_STRIDE) = pack2bf(f0.z, f1.z);
            *(unsigned*)(dst + 3 * B_STRIDE) = pack2bf(f0.w, f1.w);
        }
    };

    // Fragment lane geometry (ISA 16-bit 16x32 A layout; B mirrored):
    // lanes 0-15: M=lane, K in {0..7, 16..23}; lanes 16-31: M=lane-16, K in {8..15, 24..31}
    const int col   = lane & 15;
    const int half  = lane >> 4;
    const int khalf = half * 8;

    float scoreAcc[8];
    #pragma unroll
    for (int r = 0; r < 8; ++r) scoreAcc[r] = 0.f;

    for (int n0 = 0; n0 < U_; n0 += TN) {
        v8f c0 = {0.f, 0.f, 0.f, 0.f, 0.f, 0.f, 0.f, 0.f};
        v8f c1 = {0.f, 0.f, 0.f, 0.f, 0.f, 0.f, 0.f, 0.f};

        stageB(0, 0, n0);                 // prologue stage (safe: buf0's previous
        int buf = 0;                      // readers were fenced by the last sync)
        for (int k0 = 0; k0 < D_; k0 += TK, buf ^= 1) {
            __syncthreads();              // lB[buf] staged; lB[buf^1] readers done
            if (k0 + TK < D_)
                stageB(buf ^ 1, k0 + TK, n0);   // overlap with WMMA on lB[buf]

            Frag a, b0, b1;
            const unsigned short* pa = &lA[(waveM * 16 + col) * A_STRIDE + k0 + khalf];
            a.p.lo = *(const v8us*)(pa);        // K: khalf + 0..7
            a.p.hi = *(const v8us*)(pa + 16);   // K: khalf + 16..23
            const unsigned short* pb0 = &lB[buf][(waveN * 32 + col) * B_STRIDE + khalf];
            const unsigned short* pb1 = pb0 + 16 * B_STRIDE;
            b0.p.lo = *(const v8us*)(pb0);
            b0.p.hi = *(const v8us*)(pb0 + 16);
            b1.p.lo = *(const v8us*)(pb1);
            b1.p.hi = *(const v8us*)(pb1 + 16);

            c0 = __builtin_amdgcn_wmma_f32_16x16x32_bf16(
                     false, a.v, false, b0.v, (short)0, c0, false, false);
            c1 = __builtin_amdgcn_wmma_f32_16x16x32_bf16(
                     false, a.v, false, b1.v, (short)0, c1, false, false);
        }

        // Epilogue for this U strip, straight on the C fragments:
        // C layout: VGPR r -> row M = r (lanes 0-15) / r+8 (lanes 16-31), col N = lane&15
        const int ug0 = n0 + waveN * 32 + col;
        const int ug1 = ug0 + 16;
        const float wb0 = whbm[b * U_ + ug0];
        const float wv0 = Wv[ug0];
        const float wb1 = whbm[b * U_ + ug1];
        const float wv1 = Wv[ug1];
        #pragma unroll
        for (int r = 0; r < 8; ++r)
            scoreAcc[r] += tanhf(c0[r] + wb0) * wv0 + tanhf(c1[r] + wb1) * wv1;
    }

    // Reduce over the 16 N-columns held by each 16-lane half.
    #pragma unroll
    for (int r = 0; r < 8; ++r) {
        float v = scoreAcc[r];
        v += __shfl_xor(v, 1);
        v += __shfl_xor(v, 2);
        v += __shfl_xor(v, 4);
        v += __shfl_xor(v, 8);
        scoreAcc[r] = v;
    }
    if (tid < TM) lds_score[tid] = 0.f;
    __syncthreads();
    if (col == 0) {
        #pragma unroll
        for (int r = 0; r < 8; ++r)
            atomicAdd(&lds_score[waveM * 16 + r + half * 8], scoreAcc[r]);  // ds_add_f32
    }
    __syncthreads();
    if (tid < TM)
        scores[b * T_ + t0 + tid] = lds_score[tid];
}

// ---------------------------------------------------------------------------
// Kernel 3: softmax over T=512 per batch.
// ---------------------------------------------------------------------------
__global__ __launch_bounds__(NTHREADS) void softmax_kernel(
    const float* __restrict__ scores, float* __restrict__ at)
{
    __shared__ float red[NTHREADS];
    const int b = blockIdx.x, tid = threadIdx.x;
    float x0 = scores[b * T_ + tid];
    float x1 = scores[b * T_ + 256 + tid];
    red[tid] = fmaxf(x0, x1);
    __syncthreads();
    for (int s = 128; s > 0; s >>= 1) {
        if (tid < s) red[tid] = fmaxf(red[tid], red[tid + s]);
        __syncthreads();
    }
    const float m = red[0];
    __syncthreads();
    float e0 = __expf(x0 - m), e1 = __expf(x1 - m);
    red[tid] = e0 + e1;
    __syncthreads();
    for (int s = 128; s > 0; s >>= 1) {
        if (tid < s) red[tid] += red[tid + s];
        __syncthreads();
    }
    const float inv = 1.f / red[0];
    at[b * T_ + tid]       = e0 * inv;
    at[b * T_ + 256 + tid] = e1 * inv;
}

// ---------------------------------------------------------------------------
// Kernel 4: ct = mt * at   (pure bandwidth, float4)
// ---------------------------------------------------------------------------
__global__ __launch_bounds__(NTHREADS) void ct_kernel(
    const float* __restrict__ mt, const float* __restrict__ at,
    float* __restrict__ ct)
{
    const size_t i4 = (size_t)blockIdx.x * NTHREADS + threadIdx.x; // float4 index
    const float4* m4 = (const float4*)mt;
    float4* o4 = (float4*)ct;
    const int bt = (int)(i4 >> 8);        // D_/4 = 256 float4 per (b,t)
    const float a = at[bt];
    float4 m = m4[i4];
    float4 o; o.x = m.x * a; o.y = m.y * a; o.z = m.z * a; o.w = m.w * a;
    o4[i4] = o;
}

extern "C" void kernel_launch(void* const* d_in, const int* in_sizes, int n_in,
                              void* d_out, int out_size, void* d_ws, size_t ws_size,
                              hipStream_t stream) {
    const float* mt = (const float*)d_in[0];  // [B,T,D]
    const float* ht = (const float*)d_in[1];  // [B,1,D]
    const float* Wm = (const float*)d_in[2];  // [D,U]
    const float* bm = (const float*)d_in[3];  // [U]
    const float* Wh = (const float*)d_in[4];  // [D,U]
    const float* bh = (const float*)d_in[5];  // [U]
    const float* Wv = (const float*)d_in[6];  // [U,1]
    // d_in[7] = bv: softmax is shift-invariant -> unused.
    (void)in_sizes; (void)n_in; (void)out_size; (void)ws_size;

    float* ct = (float*)d_out;                       // [B,T,D]
    float* at = ct + (size_t)B_ * T_ * D_;           // [B,T,1]
    float* whbm   = (float*)d_ws;                    // [B,U]
    float* scores = whbm + (size_t)B_ * U_;          // [B,T]

    whbm_kernel<<<dim3(U_ / NTHREADS, B_), NTHREADS, 0, stream>>>(ht, Wh, bh, bm, whbm);
    bahdanau_scores_kernel<<<dim3(T_ / TM, B_), NTHREADS, 0, stream>>>(mt, Wm, whbm, Wv, scores);
    softmax_kernel<<<B_, NTHREADS, 0, stream>>>(scores, at);
    ct_kernel<<<(B_ * T_ * D_ / 4) / NTHREADS, NTHREADS, 0, stream>>>(mt, at, ct);
}
